// GAT_3p_uw_81063212744713
// MI455X (gfx1250) — matrix operand order repaired and checked by reading, not compiled, not run on hardware
//
#include <hip/hip_runtime.h>
#include <hip/hip_bf16.h>

// ---------------- problem constants (match reference) ----------------
#define NN      50000
#define EE      1600000
#define EP      (EE + NN)          // edges incl. self loops
#define GG      128
#define IN_F    64
#define HH      4
#define C1      32
#define HC1     (HH * C1)          // 128
#define C2      64
#define OUT_F   2
#define EPS     1e-5f
#define SLOPE   0.2f
#define NEG_BIG (-3.0e38f)

typedef __attribute__((ext_vector_type(16))) __bf16    v16bf;
typedef __attribute__((ext_vector_type(8)))  float     v8f;
typedef __attribute__((ext_vector_type(4)))  unsigned  v4u;
typedef __attribute__((ext_vector_type(8)))  unsigned  v8u;

// ---------------------------------------------------------------------
// TDM: DMA the whole W panel (K x NC f32, row-major) from global into LDS.
// D# packing per CDNA5 ISA 8.3/8.4 (2-D tensor -> groups 0 and 1 only).
// Issued by one wave; completion via s_wait_tensorcnt 0.
// ---------------------------------------------------------------------
__device__ __forceinline__ void tdm_load_w(const float* W, float* sW, int K, int NC)
{
    unsigned long long ga = (unsigned long long)(size_t)W;   // global byte addr
    unsigned lds          = (unsigned)(size_t)sW;            // LDS byte offset (low 32 bits)

    v4u g0;
    g0[0] = 1u;                                   // count=1 (valid), no gather
    g0[1] = lds;                                  // lds_addr
    g0[2] = (unsigned)(ga & 0xffffffffull);       // global_addr[31:0]
    g0[3] = (unsigned)(ga >> 32) | 0x80000000u;   // global_addr[56:32] | type=2<<30

    v8u g1;
    g1[0] = 2u << 16;                             // wg_mask=0 | data_size=2 (4 bytes)
    g1[1] = ((unsigned)NC & 0xffffu) << 16;       // tensor_dim0[15:0]  (bits 63:48)
    g1[2] = ((unsigned)K  & 0xffffu) << 16;       // dim0 hi=0 | tensor_dim1[15:0]
    g1[3] = ((unsigned)NC & 0xffffu) << 16;       // dim1 hi=0 | tile_dim0 = NC
    g1[4] = (unsigned)K & 0xffffu;                // tile_dim1 = K | tile_dim2 = 0
    g1[5] = (unsigned)NC;                         // tensor_dim0_stride[31:0]
    g1[6] = 0u;                                   // stride hi | dim1_stride lo
    g1[7] = 0u;

    asm volatile("tensor_load_to_lds %0, %1" :: "s"(g0), "s"(g1) : "memory");
    __builtin_amdgcn_s_wait_tensorcnt(0);
}

// ---------------------------------------------------------------------
// WMMA GEMM:  Out[M x NC] = A[M x K] @ W[K x NC] + bias
// K, NC compile-time -> K-loop fully unrolled (back-to-back v_wmma).
// blockDim = 2*NC (NC/16 waves); wave w owns column tile w; grid.x = M/16.
// W panel staged in LDS via TDM; A rows loaded as float4, cvt -> bf16.
// ---------------------------------------------------------------------
template <int K, int NC>
__global__ void wmma_gemm_bias(const float* __restrict__ A,
                               const float* __restrict__ W,
                               const float* __restrict__ bias,
                               float* __restrict__ Out, int M)
{
    __shared__ float sW[K * NC];                 // <= 32 KB of the 320 KB/WGP LDS

    const int lane = threadIdx.x & 31;
    const int warp = threadIdx.x >> 5;           // == column tile (all waves active)
    const int tm   = blockIdx.x;

    if (warp == 0) tdm_load_w(W, sW, K, NC);     // one wave drives the TDM
    __syncthreads();                             // publish LDS to the block

    const int hi  = lane >> 4;                   // 0 or 1
    const int lo  = lane & 15;
    const int row = tm * 16 + lo;                // A-fragment row for this lane
    const int col = warp * 16 + lo;              // B-fragment col for this lane

    v8f acc = {};
#pragma unroll
    for (int kb = 0; kb < K; kb += 32) {
        // A fragment (ISA 7.12.2, 16-bit A 16x32):
        //   lanes 0-15 : K = kb + {0..7, 16..23}
        //   lanes 16-31: K = kb + {8..15, 24..31}
        const float4* ap = (const float4*)(A + (size_t)row * K + kb + hi * 8);
        float4 a0 = ap[0];
        float4 a1 = ap[1];
        float4 a2 = ap[4];                       // +16 floats
        float4 a3 = ap[5];
        v16bf a;
        a[0]=(__bf16)a0.x;  a[1]=(__bf16)a0.y;  a[2]=(__bf16)a0.z;  a[3]=(__bf16)a0.w;
        a[4]=(__bf16)a1.x;  a[5]=(__bf16)a1.y;  a[6]=(__bf16)a1.z;  a[7]=(__bf16)a1.w;
        a[8]=(__bf16)a2.x;  a[9]=(__bf16)a2.y;  a[10]=(__bf16)a2.z; a[11]=(__bf16)a2.w;
        a[12]=(__bf16)a3.x; a[13]=(__bf16)a3.y; a[14]=(__bf16)a3.z; a[15]=(__bf16)a3.w;

        // B fragment from LDS: lane holds column `col`, element i -> K = kb + i + 8*hi + (i >= 8 ? 8 : 0)
        v16bf b;
        const float* wcol = sW + (size_t)kb * NC + col;
#pragma unroll
        for (int i = 0; i < 16; ++i) b[i] = (__bf16)wcol[(i + hi * 8 + (i >= 8 ? 8 : 0)) * NC];

        acc = __builtin_amdgcn_wmma_f32_16x16x32_bf16(
                  false, a, false, b, (short)0, acc, false, false);
    }

    // C/D layout: VGPR r -> row = r + 8*hi, col = lo
    const float bcol = bias[col];
#pragma unroll
    for (int r = 0; r < 8; ++r) {
        int orow = tm * 16 + r + 8 * hi;
        Out[(size_t)orow * NC + col] = acc[r] + bcol;
    }
}

// ---------------------------------------------------------------------
// generic fill
// ---------------------------------------------------------------------
__global__ void fill_kernel(float* __restrict__ p, float v, long long n)
{
    long long i = (long long)blockIdx.x * blockDim.x + threadIdx.x;
    long long stride = (long long)gridDim.x * blockDim.x;
    for (; i < n; i += stride) p[i] = v;
}

// ---------------------------------------------------------------------
// Layer-1 edge kernels (H=4 heads, C=32; wave per edge, lane per channel)
// ---------------------------------------------------------------------
__global__ void edge_logits1(const long long* __restrict__ ei,
                             const float* __restrict__ xl,
                             const float* __restrict__ xr,
                             const float* __restrict__ att,
                             float* __restrict__ logits,
                             float* __restrict__ maxv)
{
    int wid  = blockIdx.x * 8 + (threadIdx.x >> 5);
    int lane = threadIdx.x & 31;
    if (wid >= EP) return;
    long long s, d;
    if (wid < EE) { s = ei[wid]; d = ei[(size_t)EE + wid]; }
    else          { s = d = wid - EE; }

    const float* xs = xl + (size_t)s * HC1;
    const float* xd = xr + (size_t)d * HC1;
    __builtin_prefetch(xs, 0, 0);
    __builtin_prefetch(xd, 0, 0);

#pragma unroll
    for (int h = 0; h < HH; ++h) {
        float v = xs[h * C1 + lane] + xd[h * C1 + lane];
        v = v > 0.f ? v : SLOPE * v;                 // leaky_relu
        float t = v * att[h * C1 + lane];
#pragma unroll
        for (int off = 16; off; off >>= 1) t += __shfl_xor(t, off, 32);
        if (lane == 0) {
            logits[(size_t)wid * HH + h] = t;
            atomicMax(&maxv[(size_t)d * HH + h], t);
        }
    }
}

__global__ void edge_expsum1(const long long* __restrict__ ei,
                             float* __restrict__ logits,
                             const float* __restrict__ maxv,
                             float* __restrict__ sum)
{
    int e = blockIdx.x * blockDim.x + threadIdx.x;
    if (e >= EP) return;
    long long d = (e < EE) ? ei[(size_t)EE + e] : (long long)(e - EE);
#pragma unroll
    for (int h = 0; h < HH; ++h) {
        float l = logits[(size_t)e * HH + h];
        float w = __expf(l - maxv[(size_t)d * HH + h]);
        logits[(size_t)e * HH + h] = w;              // overwrite with exp()
        atomicAdd(&sum[(size_t)d * HH + h], w);
    }
}

__global__ void edge_scatter1(const long long* __restrict__ ei,
                              const float* __restrict__ xl,
                              const float* __restrict__ logits,
                              const float* __restrict__ sum,
                              float* __restrict__ out)
{
    int wid  = blockIdx.x * 8 + (threadIdx.x >> 5);
    int lane = threadIdx.x & 31;
    if (wid >= EP) return;
    long long s, d;
    if (wid < EE) { s = ei[wid]; d = ei[(size_t)EE + wid]; }
    else          { s = d = wid - EE; }

    const float* xs = xl + (size_t)s * HC1;
    float*       od = out + (size_t)d * HC1;
#pragma unroll
    for (int h = 0; h < HH; ++h) {
        float w = logits[(size_t)wid * HH + h] / sum[(size_t)d * HH + h];
        atomicAdd(&od[h * C1 + lane], w * xs[h * C1 + lane]);
    }
}

// ---------------------------------------------------------------------
// Layer-2 edge kernels (1 head, C=64; wave per edge, 2 channels per lane)
// ---------------------------------------------------------------------
__global__ void edge_logits2(const long long* __restrict__ ei,
                             const float* __restrict__ xl,
                             const float* __restrict__ xr,
                             const float* __restrict__ att,
                             float* __restrict__ logits,
                             float* __restrict__ maxv)
{
    int wid  = blockIdx.x * 8 + (threadIdx.x >> 5);
    int lane = threadIdx.x & 31;
    if (wid >= EP) return;
    long long s, d;
    if (wid < EE) { s = ei[wid]; d = ei[(size_t)EE + wid]; }
    else          { s = d = wid - EE; }

    const float* xs = xl + (size_t)s * C2;
    const float* xd = xr + (size_t)d * C2;
    __builtin_prefetch(xs, 0, 0);

    float v0 = xs[lane]      + xd[lane];
    float v1 = xs[lane + 32] + xd[lane + 32];
    v0 = v0 > 0.f ? v0 : SLOPE * v0;
    v1 = v1 > 0.f ? v1 : SLOPE * v1;
    float t = v0 * att[lane] + v1 * att[lane + 32];
#pragma unroll
    for (int off = 16; off; off >>= 1) t += __shfl_xor(t, off, 32);
    if (lane == 0) {
        logits[wid] = t;
        atomicMax(&maxv[d], t);
    }
}

__global__ void edge_expsum2(const long long* __restrict__ ei,
                             float* __restrict__ logits,
                             const float* __restrict__ maxv,
                             float* __restrict__ sum)
{
    int e = blockIdx.x * blockDim.x + threadIdx.x;
    if (e >= EP) return;
    long long d = (e < EE) ? ei[(size_t)EE + e] : (long long)(e - EE);
    float w = __expf(logits[e] - maxv[d]);
    logits[e] = w;
    atomicAdd(&sum[d], w);
}

__global__ void edge_scatter2(const long long* __restrict__ ei,
                              const float* __restrict__ xl,
                              const float* __restrict__ logits,
                              const float* __restrict__ sum,
                              float* __restrict__ out)
{
    int wid  = blockIdx.x * 8 + (threadIdx.x >> 5);
    int lane = threadIdx.x & 31;
    if (wid >= EP) return;
    long long s, d;
    if (wid < EE) { s = ei[wid]; d = ei[(size_t)EE + wid]; }
    else          { s = d = wid - EE; }

    const float* xs = xl + (size_t)s * C2;
    float*       od = out + (size_t)d * C2;
    float w = logits[wid] / sum[d];
    atomicAdd(&od[lane],      w * xs[lane]);
    atomicAdd(&od[lane + 32], w * xs[lane + 32]);
}

// ---------------------------------------------------------------------
// elementwise:  h = relu(h + bias[col])
// ---------------------------------------------------------------------
__global__ void bias_relu(float* __restrict__ h, const float* __restrict__ bias,
                          long long n, int cols)
{
    long long i = (long long)blockIdx.x * blockDim.x + threadIdx.x;
    if (i >= n) return;
    float v = h[i] + bias[i % cols];
    h[i] = v > 0.f ? v : 0.f;
}

// ---------------------------------------------------------------------
// batchnorm over rows (batch statistics): one block per column
// stats[c] = mean, stats[cols+c] = rsqrt(var + eps)
// ---------------------------------------------------------------------
__global__ void bn_stats(const float* __restrict__ h, float* __restrict__ stats,
                         int n, int cols)
{
    int c = blockIdx.x;
    float s = 0.f, s2 = 0.f;
    for (int i = threadIdx.x; i < n; i += blockDim.x) {
        float v = h[(size_t)i * cols + c];
        s += v; s2 += v * v;
    }
    __shared__ float sh[256], sh2[256];
    sh[threadIdx.x] = s; sh2[threadIdx.x] = s2;
    __syncthreads();
    for (int off = blockDim.x >> 1; off; off >>= 1) {
        if ((int)threadIdx.x < off) {
            sh[threadIdx.x]  += sh[threadIdx.x + off];
            sh2[threadIdx.x] += sh2[threadIdx.x + off];
        }
        __syncthreads();
    }
    if (threadIdx.x == 0) {
        float mu  = sh[0] / (float)n;
        float var = sh2[0] / (float)n - mu * mu;
        stats[c]        = mu;
        stats[cols + c] = rsqrtf(var + EPS);
    }
}

__global__ void bn_apply(float* __restrict__ h, const float* __restrict__ stats,
                         const float* __restrict__ gamma, const float* __restrict__ beta,
                         long long n, int cols)
{
    long long i = (long long)blockIdx.x * blockDim.x + threadIdx.x;
    if (i >= n) return;
    int c = (int)(i % cols);
    h[i] = (h[i] - stats[c]) * stats[cols + c] * gamma[c] + beta[c];
}

// ---------------------------------------------------------------------
// graph pooling: add / count / max per graph
// ---------------------------------------------------------------------
__global__ void pool_kernel(const float* __restrict__ h,
                            const long long* __restrict__ batch,
                            float* __restrict__ psum, float* __restrict__ pmax,
                            float* __restrict__ pcnt)
{
    long long i = (long long)blockIdx.x * blockDim.x + threadIdx.x;
    if (i >= (long long)NN * C2) return;
    int node = (int)(i >> 6);
    int c    = (int)(i & 63);
    int g    = (int)batch[node];
    float v  = h[i];
    atomicAdd(&psum[g * C2 + c], v);
    atomicMax(&pmax[g * C2 + c], v);
    if (c == 0) atomicAdd(&pcnt[g], 1.0f);
}

// feat = [sum, mean, max] (G x 192); out = feat @ lin_w + lin_b  -> [G x 2]
__global__ void final_linear(const float* __restrict__ psum,
                             const float* __restrict__ pmax,
                             const float* __restrict__ pcnt,
                             const float* __restrict__ lw,
                             const float* __restrict__ lb,
                             float* __restrict__ out)
{
    int idx = blockIdx.x * blockDim.x + threadIdx.x;
    if (idx >= GG * OUT_F) return;
    int g = idx >> 1, o = idx & 1;
    float inv = 1.0f / fmaxf(pcnt[g], 1.0f);
    float acc = lb[o];
#pragma unroll 4
    for (int j = 0; j < C2; ++j) acc += psum[g * C2 + j]       * lw[j * OUT_F + o];
#pragma unroll 4
    for (int j = 0; j < C2; ++j) acc += psum[g * C2 + j] * inv * lw[(C2 + j) * OUT_F + o];
#pragma unroll 4
    for (int j = 0; j < C2; ++j) acc += pmax[g * C2 + j]       * lw[(2 * C2 + j) * OUT_F + o];
    out[idx] = acc;
}

// ---------------------------------------------------------------------
extern "C" void kernel_launch(void* const* d_in, const int* in_sizes, int n_in,
                              void* d_out, int out_size, void* d_ws, size_t ws_size,
                              hipStream_t stream)
{
    const float*     x     = (const float*)d_in[0];
    const long long* ei    = (const long long*)d_in[1];   // int64 [2,E]
    const long long* batch = (const long long*)d_in[2];   // int64 [N]
    const float* W1l  = (const float*)d_in[3];
    const float* b1l  = (const float*)d_in[4];
    const float* W1r  = (const float*)d_in[5];
    const float* b1r  = (const float*)d_in[6];
    const float* att1 = (const float*)d_in[7];
    const float* bias1= (const float*)d_in[8];
    const float* W2l  = (const float*)d_in[9];
    const float* b2l  = (const float*)d_in[10];
    const float* W2r  = (const float*)d_in[11];
    const float* b2r  = (const float*)d_in[12];
    const float* att2 = (const float*)d_in[13];
    const float* bias2= (const float*)d_in[14];
    const float* bn1g = (const float*)d_in[15];
    const float* bn1b = (const float*)d_in[16];
    const float* bn2g = (const float*)d_in[17];
    const float* bn2b = (const float*)d_in[18];
    const float* linw = (const float*)d_in[19];
    const float* linb = (const float*)d_in[20];
    float* out = (float*)d_out;

    // ---- workspace carve-up ----
    float* ws = (float*)d_ws;
    size_t off = 0;
    auto alloc = [&](size_t n) { float* p = ws + off; off += n; return p; };
    float* xl1 = alloc((size_t)NN * HC1);
    float* xr1 = alloc((size_t)NN * HC1);
    float* h1  = alloc((size_t)NN * HC1);
    float* xl2 = alloc((size_t)NN * C2);
    float* xr2 = alloc((size_t)NN * C2);
    float* h2  = alloc((size_t)NN * C2);
    float* lg1 = alloc((size_t)EP * HH);
    float* lg2 = alloc((size_t)EP);
    float* mx1 = alloc((size_t)NN * HH);
    float* sm1 = alloc((size_t)NN * HH);
    float* mx2 = alloc((size_t)NN);
    float* sm2 = alloc((size_t)NN);
    float* st1 = alloc(2 * HC1);
    float* st2 = alloc(2 * C2);
    float* psum= alloc((size_t)GG * C2);
    float* pmax= alloc((size_t)GG * C2);
    float* pcnt= alloc((size_t)GG);
    (void)ws_size; (void)in_sizes; (void)n_in; (void)out_size;

    const int TB = 256;
    const int EW = (EP + 7) / 8;                 // wave-per-edge grids

    // ---- init accumulators (re-initialized every call: deterministic) ----
    fill_kernel<<<1024, TB, 0, stream>>>(h1,  0.f,     (long long)NN * HC1);
    fill_kernel<<<1024, TB, 0, stream>>>(h2,  0.f,     (long long)NN * C2);
    fill_kernel<<<256,  TB, 0, stream>>>(mx1, NEG_BIG, (long long)NN * HH);
    fill_kernel<<<256,  TB, 0, stream>>>(sm1, 0.f,     (long long)NN * HH);
    fill_kernel<<<128,  TB, 0, stream>>>(mx2, NEG_BIG, (long long)NN);
    fill_kernel<<<128,  TB, 0, stream>>>(sm2, 0.f,     (long long)NN);
    fill_kernel<<<32,   TB, 0, stream>>>(psum, 0.f,    (long long)GG * C2);
    fill_kernel<<<32,   TB, 0, stream>>>(pmax, NEG_BIG,(long long)GG * C2);
    fill_kernel<<<1,    TB, 0, stream>>>(pcnt, 0.f,    (long long)GG);

    // ---- layer 1 node transforms (WMMA + TDM-staged W) ----
    wmma_gemm_bias<IN_F, HC1><<<NN / 16, 2 * HC1, 0, stream>>>(x, W1l, b1l, xl1, NN);
    wmma_gemm_bias<IN_F, HC1><<<NN / 16, 2 * HC1, 0, stream>>>(x, W1r, b1r, xr1, NN);

    // ---- layer 1 edge phase: segment softmax + aggregate ----
    edge_logits1 <<<EW, TB, 0, stream>>>(ei, xl1, xr1, att1, lg1, mx1);
    edge_expsum1 <<<(EP + TB - 1) / TB, TB, 0, stream>>>(ei, lg1, mx1, sm1);
    edge_scatter1<<<EW, TB, 0, stream>>>(ei, xl1, lg1, sm1, h1);
    bias_relu<<<(int)(((long long)NN * HC1 + TB - 1) / TB), TB, 0, stream>>>(h1, bias1, (long long)NN * HC1, HC1);
    bn_stats <<<HC1, TB, 0, stream>>>(h1, st1, NN, HC1);
    bn_apply <<<(int)(((long long)NN * HC1 + TB - 1) / TB), TB, 0, stream>>>(h1, st1, bn1g, bn1b, (long long)NN * HC1, HC1);

    // ---- layer 2 node transforms (WMMA + TDM-staged W) ----
    wmma_gemm_bias<HC1, C2><<<NN / 16, 2 * C2, 0, stream>>>(h1, W2l, b2l, xl2, NN);
    wmma_gemm_bias<HC1, C2><<<NN / 16, 2 * C2, 0, stream>>>(h1, W2r, b2r, xr2, NN);

    // ---- layer 2 edge phase ----
    edge_logits2 <<<EW, TB, 0, stream>>>(ei, xl2, xr2, att2, lg2, mx2);
    edge_expsum2 <<<(EP + TB - 1) / TB, TB, 0, stream>>>(ei, lg2, mx2, sm2);
    edge_scatter2<<<EW, TB, 0, stream>>>(ei, xl2, lg2, sm2, h2);
    bias_relu<<<(int)(((long long)NN * C2 + TB - 1) / TB), TB, 0, stream>>>(h2, bias2, (long long)NN * C2, C2);
    bn_stats <<<C2, TB, 0, stream>>>(h2, st2, NN, C2);
    bn_apply <<<(int)(((long long)NN * C2 + TB - 1) / TB), TB, 0, stream>>>(h2, st2, bn2g, bn2b, (long long)NN * C2, C2);

    // ---- pooling + final linear ----
    pool_kernel<<<(int)(((long long)NN * C2 + TB - 1) / TB), TB, 0, stream>>>(h2, batch, psum, pmax, pcnt);
    final_linear<<<1, TB, 0, stream>>>(psum, pmax, pcnt, linw, linb, out);
}
